// LSTMMEM_14559939133613
// MI455X (gfx1250) — matrix operand order repaired and checked by reading, not compiled, not run on hardware
//
#include <hip/hip_runtime.h>
#include <hip/hip_bf16.h>

typedef _Float16 half_t;
typedef __attribute__((ext_vector_type(16))) _Float16 v16h;
typedef __attribute__((ext_vector_type(8)))  float    v8f;

#define H_    150
#define HP_   160     // padded hidden (recurrent K) = 5 k-steps of 32
#define G_    600
#define GP_   608     // padded gates = 38 tiles of 16
#define NTILES 38
#define KP_   320     // padded input K (emb 300 / conv 303) = 10 k-steps of 32
#define HALFG 304     // gate half staged in LDS by the gx GEMM (19 tiles)

__device__ inline float sigf(float x){ return 1.f/(1.f+__expf(-x)); }

// A fragment: 16x32 f16. lane -> row (lane&15); half-group hg = lane>>4.
// halves 0..15 -> K = kbase + 16*hg + j
__device__ inline v16h load_a_frag(const half_t* rows, int KP, int kbase, int lane){
  int m = lane & 15, hg = (lane >> 4) & 1;
  const half_t* p0 = rows + m*KP + kbase + 16*hg;
  v16h a;
#pragma unroll
  for (int j = 0; j < 16; ++j){ a[j] = p0[j]; }
  return a;
}

// B fragment: 32x16 f16 from row-major W[ncols][KP] (B[k][n] = W[n][k]).
// lane -> column n; lane group kg -> K offset {0,16}; 16 contiguous halves.
__device__ inline v16h load_b_frag(const half_t* W, int KP, int nbase, int kbase, int lane){
  int n  = nbase + (lane & 15);
  int kg = (lane >> 4) & 1;
  const half_t* p = W + (size_t)n*KP + kbase + 16*kg;
  v16h b;
#pragma unroll
  for (int h = 0; h < 16; ++h) b[h] = p[h];
  return b;
}

// ---------------------------------------------------------------------------
// Phase 1: gx[row][GP_] (f16) = x[row] @ Wih^T  batched over all (seq, t).
// Block: 256 threads (8 waves) = 128 rows x one gate-half (304 gates).
// LDS: X tile (80KB) + Wih half (190KB). Waves keep 10 A-frags in registers.
// mode 0: x = E[tok[n, t_eff]] ; mode 1: x = Xdense[rowtok]
// ---------------------------------------------------------------------------
__global__ __launch_bounds__(256) void gemm_gx_kernel(
    int mode, const int* tok, int tokRowStride, int tokOff, int tokTStride,
    const float* Xdense, const float* E, const int* lens,
    const half_t* Wih,          // [GP_][KP_] f16 padded
    half_t* gx,                 // [rows][GP_]
    int T, int Din, int dirBwd)
{
  __shared__ half_t xs[128*KP_];     // 81920 B
  __shared__ half_t wl[HALFG*KP_];   // 194560 B
  __shared__ int    rowidx[128];

  const int tid  = threadIdx.x;
  const int lane = tid & 31;
  const int wave = __builtin_amdgcn_readfirstlane(tid >> 5);  // scalar wave id
  const int r0   = blockIdx.x * 128;
  const int g0   = blockIdx.y * HALFG;

  // per-row token (mode 0) or dense-row index (mode 1)
  if (tid < 128) {
    int rg = r0 + tid;
    int n = rg / T, t = rg - n*T;
    int L = lens[n];
    int te = t;
    if (dirBwd) { te = L - 1 - t; te = te < 0 ? 0 : te; te = te > T-1 ? T-1 : te; }
    rowidx[tid] = (mode == 0)
        ? tok[(size_t)n*tokRowStride + tokOff + (size_t)te*tokTStride]
        : (n*T + te);
  }
  // stage Wih half (contiguous rows g0..g0+303) into LDS
  {
    const uint2* src = (const uint2*)(Wih + (size_t)g0*KP_);
    uint2* dst = (uint2*)wl;
    for (int i = tid; i < HALFG*KP_/4; i += 256) dst[i] = src[i];
  }
  __syncthreads();
  // stage X tile (f32 -> f16)
  for (int idx = tid; idx < 128*(KP_/4); idx += 256) {
    int r  = idx / (KP_/4);
    int c4 = (idx - r*(KP_/4)) * 4;
    float v0=0.f, v1=0.f, v2=0.f, v3=0.f;
    if (mode == 0) {
      if (c4 + 3 < Din) {
        const float4 e = *(const float4*)(E + (size_t)rowidx[r]*300 + c4);
        v0=e.x; v1=e.y; v2=e.z; v3=e.w;
      }
    } else {
      const float4 e = *(const float4*)(Xdense + (size_t)rowidx[r]*KP_ + c4);
      v0=e.x; v1=e.y; v2=e.z; v3=e.w;
    }
    half_t* xp = xs + r*KP_ + c4;
    xp[0]=(half_t)v0; xp[1]=(half_t)v1; xp[2]=(half_t)v2; xp[3]=(half_t)v3;
  }
  __syncthreads();

  // cache this wave's A fragments (16 rows x 10 k-steps)
  v16h a[10];
#pragma unroll
  for (int kb = 0; kb < 10; ++kb)
    a[kb] = load_a_frag(xs + (wave*16)*KP_, KP_, kb*32, lane);

  const int nl = lane & 15, hg = (lane >> 4) & 1;
  for (int tt = 0; tt < HALFG/16; ++tt) {
    v8f acc;
#pragma unroll
    for (int v = 0; v < 8; ++v) acc[v] = 0.f;
#pragma unroll
    for (int kb = 0; kb < 10; ++kb) {
      v16h b = load_b_frag(wl, KP_, tt*16, kb*32, lane);
      acc = __builtin_amdgcn_wmma_f32_16x16x32_f16(false, a[kb], false, b,
                                                   (short)0, acc, false, false);
    }
    const size_t baseRow = (size_t)r0 + wave*16;
#pragma unroll
    for (int v = 0; v < 8; ++v) {
      int m = v + 8*hg;
      gx[(baseRow + m)*GP_ + g0 + tt*16 + nl] = (half_t)acc[v];
    }
  }
}

// ---------------------------------------------------------------------------
// Phase 2: recurrent scan. Whole Whh (608x160 f16 = 190KB) lives in LDS for
// the full T-step scan. Block: 128 threads (4 waves) = 16 sequences.
// ---------------------------------------------------------------------------
__global__ __launch_bounds__(128) void lstm_scan_kernel(
    const half_t* gx,           // [N*T][GP_] f16 (scan order)
    const half_t* Whh,          // [GP_][HP_] f16 padded
    const float* bih, const float* bhh,   // [600]
    const int* lens,
    float* hout, int outStride, int outOff,
    int N, int T)
{
  __shared__ half_t wl[GP_*HP_];   // 194560 B
  __shared__ half_t hs[16*HP_];
  __shared__ float  cs[16*H_];
  __shared__ float  gs[16*GP_];

  const int tid  = threadIdx.x;
  const int lane = tid & 31;
  const int wave = __builtin_amdgcn_readfirstlane(tid >> 5);  // scalar wave id
  const int seq0 = blockIdx.x * 16;

  { // stage Whh
    const uint2* src = (const uint2*)Whh;
    uint2* dst = (uint2*)wl;
    for (int i = tid; i < GP_*HP_/4; i += 128) dst[i] = src[i];
  }
  for (int i = tid; i < 16*HP_; i += 128) hs[i] = (half_t)0.f;
  for (int i = tid; i < 16*H_;  i += 128) cs[i] = 0.f;
  __syncthreads();

  const int nl = lane & 15, hg = (lane >> 4) & 1;
  const int tile0 = wave * 10;
  const int ntile = (tile0 + 10 <= NTILES) ? 10 : (NTILES - tile0);  // 10,10,10,8 (scalar)

  for (int t = 0; t < T; ++t) {
    // prefetch next step's gx rows
    if (t + 1 < T && tid < 16)
      __builtin_prefetch(gx + ((size_t)(seq0 + tid)*T + (t+1))*GP_, 0, 1);

    // hoist recurrent A fragments (h) once per step
    v16h a[5];
#pragma unroll
    for (int kb = 0; kb < 5; ++kb)
      a[kb] = load_a_frag(hs, HP_, kb*32, lane);

    for (int i = 0; i < ntile; ++i) {         // scalar loop bound
      int tile = tile0 + i;
      v8f acc;
#pragma unroll
      for (int v = 0; v < 8; ++v) {           // init from precomputed gx (D layout)
        int m = v + 8*hg;
        acc[v] = (float)gx[((size_t)(seq0 + m)*T + t)*GP_ + tile*16 + nl];
      }
#pragma unroll
      for (int kb = 0; kb < 5; ++kb) {
        v16h b = load_b_frag(wl, HP_, tile*16, kb*32, lane);
        acc = __builtin_amdgcn_wmma_f32_16x16x32_f16(false, a[kb], false, b,
                                                     (short)0, acc, false, false);
      }
#pragma unroll
      for (int v = 0; v < 8; ++v)
        gs[(v + 8*hg)*GP_ + tile*16 + nl] = acc[v];
    }
    __syncthreads();

    // masked LSTM cell (i,f,g,o)
    for (int idx = tid; idx < 16*H_; idx += 128) {
      int r = idx / H_, j = idx - r*H_;
      int n = seq0 + r;
      if (t < lens[n]) {
        float gi = gs[r*GP_ + j      ] + bih[j      ] + bhh[j      ];
        float gf = gs[r*GP_ + 150 + j] + bih[150 + j] + bhh[150 + j];
        float gg = gs[r*GP_ + 300 + j] + bih[300 + j] + bhh[300 + j];
        float go = gs[r*GP_ + 450 + j] + bih[450 + j] + bhh[450 + j];
        float cn = sigf(gf)*cs[r*H_+j] + sigf(gi)*tanhf(gg);
        float hn = sigf(go)*tanhf(cn);
        cs[r*H_+j] = cn;
        hs[r*HP_+j] = (half_t)hn;
      }
    }
    __syncthreads();
  }

  for (int idx = tid; idx < 16*H_; idx += 128) {
    int r = idx / H_, j = idx - r*H_;
    hout[(size_t)(seq0 + r)*outStride + outOff + j] = (float)hs[r*HP_+j];
  }
}

// ---------------------------------------------------------------------------
// Small helper kernels
// ---------------------------------------------------------------------------
__global__ void prep_w_kernel(const float* src, half_t* dst, int nmat, int G, int K, int GP, int KP){
  size_t total = (size_t)nmat*GP*KP;
  for (size_t i = (size_t)blockIdx.x*blockDim.x + threadIdx.x; i < total;
       i += (size_t)gridDim.x*blockDim.x) {
    int kp = (int)(i % KP);
    size_t t2 = i / KP;
    int gp  = (int)(t2 % GP);
    int mat = (int)(t2 / GP);
    float v = (gp < G && kp < K) ? src[((size_t)mat*G + gp)*K + kp] : 0.f;
    dst[i] = (half_t)v;
  }
}

__global__ void lens_kernel(const int* tok, int* lens, int N, int T, int rowStride, int off, int tStride){
  int n = blockIdx.x*blockDim.x + threadIdx.x;
  if (n < N) {
    int c = 0;
    for (int t = 0; t < T; ++t)
      c += (tok[(size_t)n*rowStride + off + (size_t)t*tStride] != 0) ? 1 : 0;
    lens[n] = c;
  }
}

__global__ void feats_kernel(const float* sent_reps, const int* tc, float* feats){
  int i = blockIdx.x*blockDim.x + threadIdx.x;
  if (i < 640*KP_) {
    int k = i % KP_, n = i / KP_;
    float v = 0.f;
    if (k < 300)      v = sent_reps[n*300 + k];
    else if (k < 303) v = (float)tc[(size_t)n*53 + (k - 300)];
    feats[i] = v;
  }
}

__global__ void hop_kernel(const float* repsA, const float* repsB, float* u){
  int b = blockIdx.x;
  __shared__ float uloc[300];
  __shared__ float dots[64];
  for (int d = threadIdx.x; d < 300; d += blockDim.x) uloc[d] = u[b*300 + d];
  __syncthreads();
  for (int n = threadIdx.x; n < 50; n += blockDim.x) {
    const float* r = repsA + ((size_t)b*50 + n)*300;
    float s = 0.f;
    for (int d = 0; d < 300; ++d) s += r[d]*uloc[d];
    dots[n] = s;
  }
  __syncthreads();
  if (threadIdx.x == 0) {
    float mx = -1e30f;
    for (int n = 0; n < 50; ++n) mx = fmaxf(mx, dots[n]);
    float sm = 0.f;
    for (int n = 0; n < 50; ++n){ dots[n] = __expf(dots[n]-mx); sm += dots[n]; }
    float inv = 1.f/sm;
    for (int n = 0; n < 50; ++n) dots[n] *= inv;
  }
  __syncthreads();
  for (int d = threadIdx.x; d < 300; d += blockDim.x) {
    float acc = uloc[d];
    for (int n = 0; n < 50; ++n) acc += dots[n]*repsB[((size_t)b*50 + n)*300 + d];
    u[b*300 + d] = acc;
  }
}

__global__ void out_kernel(const float* u, const float* W, const float* bias, float* out){
  int b = blockIdx.x;
  __shared__ float red[128];
  float s = 0.f;
  for (int d = threadIdx.x; d < 300; d += 128) s += u[b*300 + d]*W[d];
  red[threadIdx.x] = s;
  __syncthreads();
  for (int off = 64; off > 0; off >>= 1) {
    if (threadIdx.x < off) red[threadIdx.x] += red[threadIdx.x + off];
    __syncthreads();
  }
  if (threadIdx.x == 0) out[b] = 1.f/(1.f + __expf(-(red[0] + bias[0])));
}

// ---------------------------------------------------------------------------
extern "C" void kernel_launch(void* const* d_in, const int* in_sizes, int n_in,
                              void* d_out, int out_size, void* d_ws, size_t ws_size,
                              hipStream_t stream) {
  const int*   tc    = (const int*)  d_in[0];   // [32,20,53]
  const int*   uh    = (const int*)  d_in[1];   // [32,50,50]
  const float* E     = (const float*)d_in[2];   // [50000,300]
  const float* A_Wih = (const float*)d_in[3];   // [3,2,600,300]
  const float* A_Whh = (const float*)d_in[4];   // [3,2,600,150]
  const float* A_bih = (const float*)d_in[5];   // [3,2,600]
  const float* A_bhh = (const float*)d_in[6];
  const float* C_Wih = (const float*)d_in[7];   // [2,600,303]
  const float* C_Whh = (const float*)d_in[8];   // [2,600,150]
  const float* C_bih = (const float*)d_in[9];   // [2,600]
  const float* C_bhh = (const float*)d_in[10];
  const float* W_out = (const float*)d_in[11];  // [300]
  const float* b_out = (const float*)d_in[12];  // scalar
  float* out = (float*)d_out;
  (void)in_sizes; (void)n_in; (void)out_size; (void)ws_size;

  char* wsb = (char*)d_ws; size_t off = 0;
  auto alloc = [&](size_t bytes)->char* {
    char* p = wsb + off; off = (off + bytes + 255) & ~(size_t)255; return p;
  };
  half_t* gxbuf = (half_t*)alloc((size_t)80000*GP_*sizeof(half_t));  // 97.3 MB, reused per job
  half_t* wAih  = (half_t*)alloc((size_t)6*GP_*KP_*sizeof(half_t));
  half_t* wAhh  = (half_t*)alloc((size_t)6*GP_*HP_*sizeof(half_t));
  half_t* wCih  = (half_t*)alloc((size_t)2*GP_*KP_*sizeof(half_t));
  half_t* wChh  = (half_t*)alloc((size_t)2*GP_*HP_*sizeof(half_t));
  float*  sent  = (float*)alloc((size_t)640*300*sizeof(float));
  float*  feats = (float*)alloc((size_t)640*KP_*sizeof(float));
  float*  reps  = (float*)alloc((size_t)3*1600*300*sizeof(float));
  float*  u     = (float*)alloc((size_t)32*300*sizeof(float));
  int* slens = (int*)alloc(640*sizeof(int));
  int* hlens = (int*)alloc(1600*sizeof(int));
  int* clens = (int*)alloc(32*sizeof(int));

  // weight prep (f16, zero-padded)
  prep_w_kernel<<<256,256,0,stream>>>(A_Wih, wAih, 6, 600, 300, GP_, KP_);
  prep_w_kernel<<<256,256,0,stream>>>(A_Whh, wAhh, 6, 600, 150, GP_, HP_);
  prep_w_kernel<<<64, 256,0,stream>>>(C_Wih, wCih, 2, 600, 303, GP_, KP_);
  prep_w_kernel<<<64, 256,0,stream>>>(C_Whh, wChh, 2, 600, 150, GP_, HP_);

  // sequence lengths
  lens_kernel<<<5, 128,0,stream>>>(tc, slens,  640, 50, 53,    3, 1);
  lens_kernel<<<13,128,0,stream>>>(uh, hlens, 1600, 50, 50,    0, 1);
  lens_kernel<<<1, 128,0,stream>>>(tc, clens,   32, 20, 20*53, 3, 53);

  // sentence encoder (A enc 0) -> sent[640,300]
  for (int dir = 0; dir < 2; ++dir) {
    gemm_gx_kernel<<<dim3(250,2),256,0,stream>>>(0, tc, 53, 3, 1, nullptr, E, slens,
        wAih + (size_t)dir*GP_*KP_, gxbuf, 50, 300, dir);
    lstm_scan_kernel<<<40,128,0,stream>>>(gxbuf, wAhh + (size_t)dir*GP_*HP_,
        A_bih + dir*600, A_bhh + dir*600, slens, sent, 300, dir*150, 640, 50);
  }

  feats_kernel<<<(640*KP_+255)/256,256,0,stream>>>(sent, tc, feats);

  // conversation encoder -> u[32,300]
  for (int dir = 0; dir < 2; ++dir) {
    gemm_gx_kernel<<<dim3(5,2),256,0,stream>>>(1, nullptr, 0, 0, 0, feats, E, clens,
        wCih + (size_t)dir*GP_*KP_, gxbuf, 20, 303, dir);
    lstm_scan_kernel<<<2,128,0,stream>>>(gxbuf, wChh + (size_t)dir*GP_*HP_,
        C_bih + dir*600, C_bhh + dir*600, clens, u, 300, dir*150, 32, 20);
  }

  // history encoders k=0..2 -> reps[k][1600,300]
  for (int k = 0; k < 3; ++k)
    for (int dir = 0; dir < 2; ++dir) {
      gemm_gx_kernel<<<dim3(625,2),256,0,stream>>>(0, uh, 50, 0, 1, nullptr, E, hlens,
          wAih + (size_t)(k*2+dir)*GP_*KP_, gxbuf, 50, 300, dir);
      lstm_scan_kernel<<<100,128,0,stream>>>(gxbuf, wAhh + (size_t)(k*2+dir)*GP_*HP_,
          A_bih + (k*2+dir)*600, A_bhh + (k*2+dir)*600, hlens,
          reps + (size_t)k*1600*300, 300, dir*150, 1600, 50);
    }

  // memory hops + readout
  hop_kernel<<<32,64,0,stream>>>(reps,                    reps + (size_t)1600*300,   u);
  hop_kernel<<<32,64,0,stream>>>(reps + (size_t)1600*300, reps + (size_t)2*1600*300, u);
  out_kernel<<<32,128,0,stream>>>(u, W_out, b_out, out);
}
